// BertCRFModel_42915313221679
// MI455X (gfx1250) — matrix operand order repaired and checked by reading, not compiled, not run on hardware
//
#include <hip/hip_runtime.h>

// ---------------- model constants ----------------
#define Bc   16
#define Sc   256
#define Vc   30522
#define Hc   768
#define NHc  12
#define DHc  64
#define DFFc 3072
#define Lc   12
#define Tc   9
#define Mc   (Bc * Sc)   // 4096 tokens
#define LN_EPS 1e-12f

typedef __attribute__((ext_vector_type(16))) __bf16 v16bf;
typedef __attribute__((ext_vector_type(8)))  __bf16 v8bf;
typedef __attribute__((ext_vector_type(8)))  float  v8f;

#define WMMA_BF16(a, b, c) \
  __builtin_amdgcn_wmma_f32_16x16x32_bf16(false, (a), false, (b), (short)0, (c), false, false)

// =====================================================================
// Batched WMMA GEMM, 16(M) x 64(N) tile per wave (1 A frag x 4 B frags,
// 4 fp32 accumulators -> 4 v_wmma per 32-wide K step), software-pipelined:
// fragments for k+32 are loaded before the WMMAs for k are issued.
//   C[m][n] = act( sum_k A[m][k] * Bt[n][k] + bias[n] )
// A:  bf16 row-major MxK (lda), Bt: bf16 column-major weights NxK (ldb).
// pack: 0 plain (fp32 C [ldc] and/or bf16 Cb [ldcb])
//       1 bf16 head-pack      Cb[((b*NH+hh)*S+s)*DH+dh]   (Q,K)
//       2 bf16 head-pack-T    Cb[((b*NH+hh)*DH+dh)*S+s]   (V)
//       3 bf16 token-major    Cb[((b*S+m)*H + hh*DH + n]  (ctx, batch=b*NH+hh)
// blockIdx.y = batch. 8 waves / block.
// =====================================================================
__global__ __launch_bounds__(256) void wmma_gemm(
    const __bf16* __restrict__ A, int lda, long long sA,
    const __bf16* __restrict__ Bt, int ldb, long long sB,
    float* __restrict__ C, int ldc, long long sC,
    const float* __restrict__ bias,
    __bf16* __restrict__ Cb, int ldcb,
    int M, int N, int K, int act, int pack)
{
  const int batch = blockIdx.y;
  A  += (long long)batch * sA;
  Bt += (long long)batch * sB;
  if (C) C += (long long)batch * sC;

  const int wave = threadIdx.x >> 5;
  const int lane = threadIdx.x & 31;
  const int tilesN = N >> 6;                 // 64-wide N tiles
  const int tile = blockIdx.x * 8 + wave;
  const int tm = (tile / tilesN) << 4;
  const int tn = (tile % tilesN) << 6;
  if (tm >= M) return;   // wave-uniform: EXEC stays all-ones for WMMA

  const int half = lane >> 4;   // 0: lanes 0-15, 1: lanes 16-31
  const int l15  = lane & 15;

  // A fragment: lane holds row M=tm+l15; K = half*16 + [0..15]
  const __bf16* arow = A + (long long)(tm + l15) * lda + half * 16;
  // B fragments: lane holds column N=tn+nn*16+l15; K = half*16 + [0..15]
  const __bf16* brow0 = Bt + (long long)(tn +  0 + l15) * ldb + half * 16;
  const __bf16* brow1 = Bt + (long long)(tn + 16 + l15) * ldb + half * 16;
  const __bf16* brow2 = Bt + (long long)(tn + 32 + l15) * ldb + half * 16;
  const __bf16* brow3 = Bt + (long long)(tn + 48 + l15) * ldb + half * 16;

  v8f acc0 = {}, acc1 = {}, acc2 = {}, acc3 = {};

  // ---- prologue: fragments for k = 0 ----
  v16bf a_c, b0_c, b1_c, b2_c, b3_c;
  {
    v8bf a0 = *(const v8bf*)(arow);
    v8bf a1 = *(const v8bf*)(arow + 8);
    a_c  = __builtin_shufflevector(a0, a1,
        0,1,2,3,4,5,6,7,8,9,10,11,12,13,14,15);
    b0_c = *(const v16bf*)(brow0);
    b1_c = *(const v16bf*)(brow1);
    b2_c = *(const v16bf*)(brow2);
    b3_c = *(const v16bf*)(brow3);
  }

  for (int k = 0; k < K; k += 32) {
    // ---- issue next iteration's loads before this iteration's WMMAs ----
    v16bf a_n = a_c, b0_n = b0_c, b1_n = b1_c, b2_n = b2_c, b3_n = b3_c;
    const int kn = k + 32;
    if (kn < K) {
      if (kn + 512 < K) {   // stream-ahead prefetch (global_prefetch_b8)
        __builtin_prefetch((const void*)(arow  + kn + 512), 0, 1);
        __builtin_prefetch((const void*)(brow0 + kn + 512), 0, 1);
      }
      v8bf a0 = *(const v8bf*)(arow + kn);
      v8bf a1 = *(const v8bf*)(arow + kn + 8);
      a_n  = __builtin_shufflevector(a0, a1,
          0,1,2,3,4,5,6,7,8,9,10,11,12,13,14,15);
      b0_n = *(const v16bf*)(brow0 + kn);
      b1_n = *(const v16bf*)(brow1 + kn);
      b2_n = *(const v16bf*)(brow2 + kn);
      b3_n = *(const v16bf*)(brow3 + kn);
    }
    acc0 = WMMA_BF16(a_c, b0_c, acc0);
    acc1 = WMMA_BF16(a_c, b1_c, acc1);
    acc2 = WMMA_BF16(a_c, b2_c, acc2);
    acc3 = WMMA_BF16(a_c, b3_c, acc3);
    a_c = a_n; b0_c = b0_n; b1_c = b1_n; b2_c = b2_n; b3_c = b3_n;
  }

  const int bofs = batch / NHc;   // used by pack==3
  const int hofs = batch % NHc;
  #pragma unroll
  for (int nn = 0; nn < 4; ++nn) {
    v8f acc = (nn == 0) ? acc0 : (nn == 1) ? acc1 : (nn == 2) ? acc2 : acc3;
    const int n = tn + nn * 16 + l15;
    const float bi = bias ? bias[n] : 0.0f;
    #pragma unroll
    for (int r = 0; r < 8; ++r) {
      const int m = tm + half * 8 + r;
      float v = acc[r] + bi;
      if (act == 1) v = 0.5f * v * (1.0f + erff(v * 0.70710678118654752f)); // exact GELU
      if (pack == 0) {
        if (C)  C[(long long)m * ldc + n] = v;
        if (Cb) Cb[(long long)m * ldcb + n] = (__bf16)v;
      } else if (pack == 1) {          // head-pack (Q,K): m = b*256+s, n = hh*64+dh
        const int b_ = m >> 8, s_ = m & 255, hh = n >> 6, dh = n & 63;
        Cb[((long long)(b_ * NHc + hh) * Sc + s_) * DHc + dh] = (__bf16)v;
      } else if (pack == 2) {          // head-pack transposed (V)
        const int b_ = m >> 8, s_ = m & 255, hh = n >> 6, dh = n & 63;
        Cb[((long long)(b_ * NHc + hh) * DHc + dh) * Sc + s_] = (__bf16)v;
      } else {                          // pack == 3: ctx -> token-major bf16
        Cb[((long long)(bofs * Sc + m)) * Hc + hofs * DHc + n] = (__bf16)v;
      }
    }
  }
}

// ---------------- weight transpose + fp32 -> bf16 ----------------
// W: [K][N] fp32 row-major ->  Wt: [N][K] bf16
__global__ __launch_bounds__(256) void convert_wt(
    const float* __restrict__ W, __bf16* __restrict__ Wt, int K, int N)
{
  long long total = (long long)K * N;
  for (long long i = (long long)blockIdx.x * blockDim.x + threadIdx.x;
       i < total; i += (long long)gridDim.x * blockDim.x) {
    int n = (int)(i % N);
    long long k = i / N;
    Wt[(long long)n * K + k] = (__bf16)W[i];
  }
}

// ---------------- embedding gather + LayerNorm ----------------
__global__ __launch_bounds__(256) void embed_ln(
    const int* __restrict__ x, const float* __restrict__ we,
    const float* __restrict__ pe, const float* __restrict__ te,
    const float* __restrict__ g, const float* __restrict__ bta,
    float* __restrict__ h, __bf16* __restrict__ hb)
{
  __shared__ float red[256], red2[256];
  const int tok = blockIdx.x;
  const int s = tok % Sc;
  const int tid = threadIdx.x;
  const float* wrow = we + (long long)x[tok] * Hc;
  float e[3];
  float lsum = 0.f, lsq = 0.f;
  #pragma unroll
  for (int j = 0; j < 3; ++j) {
    int c = tid + j * 256;
    float v = wrow[c] + pe[s * Hc + c] + te[c];
    e[j] = v; lsum += v; lsq += v * v;
  }
  red[tid] = lsum; red2[tid] = lsq;
  __syncthreads();
  for (int st = 128; st > 0; st >>= 1) {
    if (tid < st) { red[tid] += red[tid + st]; red2[tid] += red2[tid + st]; }
    __syncthreads();
  }
  float mean = red[0] * (1.0f / Hc);
  float var  = red2[0] * (1.0f / Hc) - mean * mean;
  float inv  = rsqrtf(var + LN_EPS);
  #pragma unroll
  for (int j = 0; j < 3; ++j) {
    int c = tid + j * 256;
    float v = (e[j] - mean) * inv * g[c] + bta[c];
    h[(long long)tok * Hc + c] = v;
    hb[(long long)tok * Hc + c] = (__bf16)v;
  }
}

// ---------------- residual add + LayerNorm ----------------
__global__ __launch_bounds__(256) void residual_ln(
    float* __restrict__ h, const float* __restrict__ t,
    const float* __restrict__ g, const float* __restrict__ b,
    __bf16* __restrict__ hb)
{
  __shared__ float red[256], red2[256];
  const int tok = blockIdx.x;
  const int tid = threadIdx.x;
  float e[3];
  float lsum = 0.f, lsq = 0.f;
  #pragma unroll
  for (int j = 0; j < 3; ++j) {
    int c = tid + j * 256;
    float v = h[(long long)tok * Hc + c] + t[(long long)tok * Hc + c];
    e[j] = v; lsum += v; lsq += v * v;
  }
  red[tid] = lsum; red2[tid] = lsq;
  __syncthreads();
  for (int st = 128; st > 0; st >>= 1) {
    if (tid < st) { red[tid] += red[tid + st]; red2[tid] += red2[tid + st]; }
    __syncthreads();
  }
  float mean = red[0] * (1.0f / Hc);
  float var  = red2[0] * (1.0f / Hc) - mean * mean;
  float inv  = rsqrtf(var + LN_EPS);
  #pragma unroll
  for (int j = 0; j < 3; ++j) {
    int c = tid + j * 256;
    float v = (e[j] - mean) * inv * g[c] + b[c];
    h[(long long)tok * Hc + c] = v;
    hb[(long long)tok * Hc + c] = (__bf16)v;
  }
}

// ---------------- row softmax (scale then softmax over 256 cols) ----------------
__global__ __launch_bounds__(256) void softmax_rows(
    const float* __restrict__ Sc_, __bf16* __restrict__ P, float scale)
{
  const int row  = blockIdx.x * 8 + (threadIdx.x >> 5);
  const int lane = threadIdx.x & 31;
  const float* r = Sc_ + (long long)row * Sc;
  float v[8];
  float mx = -3.4e38f;
  #pragma unroll
  for (int j = 0; j < 8; ++j) {
    v[j] = r[lane + j * 32] * scale;
    mx = fmaxf(mx, v[j]);
  }
  for (int o = 16; o; o >>= 1) mx = fmaxf(mx, __shfl_xor(mx, o, 32));
  float sum = 0.f;
  #pragma unroll
  for (int j = 0; j < 8; ++j) { v[j] = __expf(v[j] - mx); sum += v[j]; }
  for (int o = 16; o; o >>= 1) sum += __shfl_xor(sum, o, 32);
  float inv = 1.0f / sum;
  #pragma unroll
  for (int j = 0; j < 8; ++j)
    P[(long long)row * Sc + lane + j * 32] = (__bf16)(v[j] * inv);
}

// ---------------- classifier: logits = h @ clf_W + clf_b  (768 -> 9) ----------------
__global__ __launch_bounds__(256) void clf_kernel(
    const float* __restrict__ h, const float* __restrict__ W,
    const float* __restrict__ b, float* __restrict__ logits)
{
  long long i = (long long)blockIdx.x * blockDim.x + threadIdx.x;
  if (i >= (long long)Mc * Tc) return;
  int t = (int)(i % Tc);
  long long m = i / Tc;
  float s = b[t];
  for (int k = 0; k < Hc; ++k) s += h[m * Hc + k] * W[k * Tc + t];
  logits[i] = s;
}

// ---------------- CRF negative mean log-likelihood ----------------
// One wave per batch element; lanes 0..8 carry the 9-tag alpha vector.
__global__ __launch_bounds__(512) void crf_loss(
    const float* __restrict__ logits, const int* __restrict__ target,
    const float* __restrict__ start, const float* __restrict__ trans,
    const float* __restrict__ endv, float* __restrict__ out)
{
  __shared__ float sllh[Bc];
  const int w = threadIdx.x >> 5;
  const int lane = threadIdx.x & 31;
  if (w < Bc) {
    const float* lg = logits + (long long)w * Sc * Tc;
    const int* tg = target + w * Sc;
    const bool act = lane < Tc;
    const int j = act ? lane : 0;
    float tr[Tc];
    #pragma unroll
    for (int i = 0; i < Tc; ++i) tr[i] = trans[i * Tc + j];
    float alpha = act ? (start[j] + lg[j]) : -1e30f;
    for (int t = 1; t < Sc; ++t) {
      bool m = tg[t] > -1;
      float av[Tc];
      float mx = -1e30f;
      #pragma unroll
      for (int i = 0; i < Tc; ++i) {
        float ai = __shfl(alpha, i, 32);
        av[i] = ai + tr[i];
        mx = fmaxf(mx, av[i]);
      }
      float s = 0.f;
      #pragma unroll
      for (int i = 0; i < Tc; ++i) s += __expf(av[i] - mx);
      float nxt = mx + __logf(s) + lg[t * Tc + j];
      if (m && act) alpha = nxt;
    }
    float aj = act ? (alpha + endv[j]) : -1e30f;
    float mx = aj;
    for (int o = 16; o; o >>= 1) mx = fmaxf(mx, __shfl_xor(mx, o, 32));
    float es = act ? __expf(aj - mx) : 0.f;
    for (int o = 16; o; o >>= 1) es += __shfl_xor(es, o, 32);
    float denom = mx + __logf(es);
    if (lane == 0) {
      int tag0 = tg[0] > -1 ? tg[0] : 0;
      float num = start[tag0] + lg[tag0];
      int cnt = (tg[0] > -1) ? 1 : 0;
      for (int t = 1; t < Sc; ++t) {
        bool m = tg[t] > -1;
        int cur  = m ? tg[t] : 0;
        int prev = tg[t - 1] > -1 ? tg[t - 1] : 0;
        if (m) { num += trans[prev * Tc + cur] + lg[t * Tc + cur]; cnt++; }
      }
      int lastIdx = cnt - 1; if (lastIdx < 0) lastIdx = 0;
      int tagLast = tg[lastIdx] > -1 ? tg[lastIdx] : 0;
      num += endv[tagLast];
      sllh[w] = num - denom;
    }
  }
  __syncthreads();
  if (threadIdx.x == 0) {
    float s = 0.f;
    for (int i = 0; i < Bc; ++i) s += sllh[i];
    out[0] = -s / Bc;
  }
}

// =====================================================================
extern "C" void kernel_launch(void* const* d_in, const int* in_sizes, int n_in,
                              void* d_out, int out_size, void* d_ws, size_t ws_size,
                              hipStream_t stream)
{
  (void)in_sizes; (void)n_in; (void)out_size; (void)ws_size;
  const int*   x        = (const int*)  d_in[0];
  const int*   target   = (const int*)  d_in[1];
  const float* word_emb = (const float*)d_in[2];
  const float* pos_emb  = (const float*)d_in[3];
  const float* type_emb = (const float*)d_in[4];
  const float* emb_g    = (const float*)d_in[5];
  const float* emb_b    = (const float*)d_in[6];
  const float* Wq  = (const float*)d_in[7];
  const float* bq  = (const float*)d_in[8];
  const float* Wk  = (const float*)d_in[9];
  const float* bk  = (const float*)d_in[10];
  const float* Wv  = (const float*)d_in[11];
  const float* bv  = (const float*)d_in[12];
  const float* Wo  = (const float*)d_in[13];
  const float* bo  = (const float*)d_in[14];
  const float* ln1g = (const float*)d_in[15];
  const float* ln1b = (const float*)d_in[16];
  const float* W1  = (const float*)d_in[17];
  const float* b1  = (const float*)d_in[18];
  const float* W2  = (const float*)d_in[19];
  const float* b2  = (const float*)d_in[20];
  const float* ln2g = (const float*)d_in[21];
  const float* ln2b = (const float*)d_in[22];
  const float* clfW = (const float*)d_in[23];
  const float* clfb = (const float*)d_in[24];
  const float* crs = (const float*)d_in[25];
  const float* crt = (const float*)d_in[26];
  const float* cre = (const float*)d_in[27];

  char* base = (char*)d_ws;
  size_t off = 0;
  auto alloc = [&](size_t bytes) -> void* {
    void* p = base + off;
    off += (bytes + 255) & ~(size_t)255;
    return p;
  };
  float*  h      = (float*) alloc((size_t)Mc * Hc * 4);
  __bf16* hbf    = (__bf16*)alloc((size_t)Mc * Hc * 2);
  __bf16* wbuf   = (__bf16*)alloc((size_t)Hc * DFFc * 2);     // per-layer bf16 weight (transposed)
  float*  tmp    = (float*) alloc((size_t)Mc * Hc * 4);       // fp32 GEMM output before residual LN
  __bf16* qbf    = (__bf16*)alloc((size_t)Mc * Hc * 2);       // [b][hh][s][dh]
  __bf16* kbf    = (__bf16*)alloc((size_t)Mc * Hc * 2);       // [b][hh][s][dh]
  __bf16* vtbf   = (__bf16*)alloc((size_t)Mc * Hc * 2);       // [b][hh][dh][s]
  float*  scores = (float*) alloc((size_t)Bc * NHc * Sc * Sc * 4);
  __bf16* probs  = (__bf16*)alloc((size_t)Bc * NHc * Sc * Sc * 2);
  __bf16* abuf   = (__bf16*)alloc((size_t)Mc * DFFc * 2);     // bf16 activations (ctx / ffn)
  float*  logits = (float*) alloc((size_t)Mc * Tc * 4);

  embed_ln<<<Mc, 256, 0, stream>>>(x, word_emb, pos_emb, type_emb, emb_g, emb_b, h, hbf);

  const long long HH = (long long)Hc * Hc;
  for (int l = 0; l < Lc; ++l) {
    // ---- Q, K, V projections (bias fused, head-packed bf16 epilogue) ----
    convert_wt<<<2304, 256, 0, stream>>>(Wq + (size_t)l * HH, wbuf, Hc, Hc);
    wmma_gemm<<<dim3(384, 1), 256, 0, stream>>>(hbf, Hc, 0, wbuf, Hc, 0,
        nullptr, 0, 0, bq + (size_t)l * Hc, qbf, 0, Mc, Hc, Hc, 0, 1);
    convert_wt<<<2304, 256, 0, stream>>>(Wk + (size_t)l * HH, wbuf, Hc, Hc);
    wmma_gemm<<<dim3(384, 1), 256, 0, stream>>>(hbf, Hc, 0, wbuf, Hc, 0,
        nullptr, 0, 0, bk + (size_t)l * Hc, kbf, 0, Mc, Hc, Hc, 0, 1);
    convert_wt<<<2304, 256, 0, stream>>>(Wv + (size_t)l * HH, wbuf, Hc, Hc);
    wmma_gemm<<<dim3(384, 1), 256, 0, stream>>>(hbf, Hc, 0, wbuf, Hc, 0,
        nullptr, 0, 0, bv + (size_t)l * Hc, vtbf, 0, Mc, Hc, Hc, 0, 2);

    // ---- attention scores: Q(SxDH) @ K^T, batched over B*NH ----
    wmma_gemm<<<dim3(8, Bc * NHc), 256, 0, stream>>>(
        qbf, DHc, (long long)Sc * DHc, kbf, DHc, (long long)Sc * DHc,
        scores, Sc, (long long)Sc * Sc, nullptr, nullptr, 0,
        Sc, Sc, DHc, 0, 0);
    softmax_rows<<<(Bc * NHc * Sc) / 8, 256, 0, stream>>>(scores, probs, 0.125f);

    // ---- ctx = P(SxS) @ V(SxDH), written token-major bf16 directly ----
    wmma_gemm<<<dim3(2, Bc * NHc), 256, 0, stream>>>(
        probs, Sc, (long long)Sc * Sc, vtbf, Sc, (long long)DHc * Sc,
        nullptr, 0, 0, nullptr, abuf, 0,
        Sc, DHc, Sc, 0, 3);

    // ---- output projection + residual LN ----
    convert_wt<<<2304, 256, 0, stream>>>(Wo + (size_t)l * HH, wbuf, Hc, Hc);
    wmma_gemm<<<dim3(384, 1), 256, 0, stream>>>(abuf, Hc, 0, wbuf, Hc, 0,
        tmp, Hc, 0, bo + (size_t)l * Hc, nullptr, 0, Mc, Hc, Hc, 0, 0);
    residual_ln<<<Mc, 256, 0, stream>>>(h, tmp, ln1g + (size_t)l * Hc, ln1b + (size_t)l * Hc, hbf);

    // ---- FFN: gelu(h @ W1 + b1) @ W2 + b2, residual LN ----
    convert_wt<<<9216, 256, 0, stream>>>(W1 + (size_t)l * Hc * DFFc, wbuf, Hc, DFFc);
    wmma_gemm<<<dim3(1536, 1), 256, 0, stream>>>(hbf, Hc, 0, wbuf, Hc, 0,
        nullptr, 0, 0, b1 + (size_t)l * DFFc, abuf, DFFc, Mc, DFFc, Hc, 1, 0);
    convert_wt<<<9216, 256, 0, stream>>>(W2 + (size_t)l * Hc * DFFc, wbuf, DFFc, Hc);
    wmma_gemm<<<dim3(384, 1), 256, 0, stream>>>(abuf, DFFc, 0, wbuf, DFFc, 0,
        tmp, Hc, 0, b2 + (size_t)l * Hc, nullptr, 0, Mc, Hc, DFFc, 0, 0);
    residual_ln<<<Mc, 256, 0, stream>>>(h, tmp, ln2g + (size_t)l * Hc, ln2b + (size_t)l * Hc, hbf);
  }

  clf_kernel<<<(Mc * Tc + 255) / 256, 256, 0, stream>>>(h, clfW, clfb, logits);
  crf_loss<<<1, 512, 0, stream>>>(logits, target, crs, crt, cre, (float*)d_out);
}
